// RNN_model_19018115187449
// MI455X (gfx1250) — compile-verified, matrix operand order checked by reading, not run on hardware
//
#include <hip/hip_runtime.h>
#include <hip/hip_bf16.h>
#include <math.h>

#define H   512
#define GN  2048   // 4*H gates
#define BB  200    // batch
#define BP  208    // batch padded to 13*16 for WMMA M-tiles
#define TT  128
#define EPS 1e-5f

typedef __attribute__((ext_vector_type(16))) __bf16 v16bf;
typedef __attribute__((ext_vector_type(8)))  __bf16 v8bf;
typedef __attribute__((ext_vector_type(8)))  float  v8f;

// Load a 16x32 bf16 WMMA fragment (A layout; B uses the same shape with rows=N columns).
// Lane L (l=L&15, hi=L>>4) holds row r0+l, K elements:
//   e=0..7  -> k0 + hi*8 + e
//   e=8..15 -> k0 + 16 + hi*8 + (e-8)
// Each half is 8 contiguous bf16 = one 16-byte (b128) load.
__device__ __forceinline__ v16bf load_frag(const __bf16* __restrict__ base, int ld,
                                           int r0, int k0, int lane) {
  int l  = lane & 15;
  int hi = lane >> 4;
  const __bf16* p = base + (size_t)(r0 + l) * ld + (k0 + hi * 8);
  v8bf lo = *(const v8bf*)p;
  v8bf hh = *(const v8bf*)(p + 16);
  return __builtin_shufflevector(lo, hh, 0,1,2,3,4,5,6,7,8,9,10,11,12,13,14,15);
}

// gates[BP][GN] = Ain[BP][H] * Wih^T + Hprev[BP][H] * Whh^T + (bih + bhh)
// Wih/Whh row-major [GN][H] bf16 -> B-matrix fragment reads contiguous weight rows.
// grid = (2, GN/16), block = 256 (8 waves). Waves in a WG share n0 (weight L0 reuse),
// wave w takes M-tile blockIdx.x*8+w (13 valid tiles).
// Two independent accumulators (input GEMM / recurrent GEMM) for XDL ILP, plus
// register double-buffering: next K-block's fragments are loaded before the
// current WMMAs issue, keeping loads in flight across the matrix ops.
__global__ __launch_bounds__(256) void lstm_gates_wmma(
    const __bf16* __restrict__ Ain,  const __bf16* __restrict__ Wih,
    const __bf16* __restrict__ Hprev,const __bf16* __restrict__ Whh,
    const float* __restrict__ bih,   const float* __restrict__ bhh,
    float* __restrict__ gbuf)
{
  int wave = threadIdx.x >> 5;
  int lane = threadIdx.x & 31;
  int mt = blockIdx.x * 8 + wave;
  if (mt >= BP / 16) return;            // whole-wave early-out (EXEC stays all-ones for WMMA)
  int m0 = mt * 16;
  int n0 = blockIdx.y * 16;

  v8f acc1 = {};
  v8f acc2 = {};

  // prologue: first K-block fragments
  v16bf a1 = load_frag(Ain,   H, m0, 0, lane);
  v16bf b1 = load_frag(Wih,   H, n0, 0, lane);
  v16bf a2 = load_frag(Hprev, H, m0, 0, lane);
  v16bf b2 = load_frag(Whh,   H, n0, 0, lane);

#pragma unroll 4
  for (int k0 = 0; k0 < H; k0 += 32) {
    // prefetch weights two K-blocks ahead (register pipeline covers +32)
    if (k0 + 64 < H) {
      __builtin_prefetch(Wih + (size_t)(n0 + (lane & 15)) * H + k0 + 64, 0, 3);
      __builtin_prefetch(Whh + (size_t)(n0 + (lane & 15)) * H + k0 + 64, 0, 3);
    }
    // issue next K-block loads BEFORE consuming current fragments
    int kn = (k0 + 32 < H) ? (k0 + 32) : 0;   // last iter: harmless reload of block 0
    v16bf na1 = load_frag(Ain,   H, m0, kn, lane);
    v16bf nb1 = load_frag(Wih,   H, n0, kn, lane);
    v16bf na2 = load_frag(Hprev, H, m0, kn, lane);
    v16bf nb2 = load_frag(Whh,   H, n0, kn, lane);

    acc1 = __builtin_amdgcn_wmma_f32_16x16x32_bf16(false, a1, false, b1, (short)0, acc1, false, false);
    acc2 = __builtin_amdgcn_wmma_f32_16x16x32_bf16(false, a2, false, b2, (short)0, acc2, false, false);

    a1 = na1; b1 = nb1; a2 = na2; b2 = nb2;
  }
  v8f acc = acc1 + acc2;

  int l = lane & 15, hi = lane >> 4;
  int n = n0 + l;
  float bias = bih[n] + bhh[n];
#pragma unroll
  for (int r = 0; r < 8; ++r)           // D layout: m = m0 + hi*8 + r, n = n0 + l
    gbuf[(size_t)(m0 + hi * 8 + r) * GN + n] = acc[r] + bias;
}

// Fused LSTM pointwise + training-mode BatchNorm + locked dropout (+ time max-pool for L3).
// One block per feature f (H blocks), 64 threads loop the batch.
__global__ __launch_bounds__(64) void lstm_cell_bn(
    const float* __restrict__ gbuf, float* __restrict__ c,
    __bf16* __restrict__ hbf, __bf16* __restrict__ abf,
    const float* __restrict__ gamma, const float* __restrict__ beta,
    const float* __restrict__ mask, float* __restrict__ pooled, int isL3)
{
  int f = blockIdx.x;
  int tid = threadIdx.x;
  __shared__ float ssum[64], ssq[64];
  float hv[4];
  float s = 0.f, sq = 0.f;
  int cnt = 0;
  for (int b = tid; b < BB; b += 64) {
    float gi = gbuf[(size_t)b * GN + f];
    float gf = gbuf[(size_t)b * GN + f + H];
    float gg = gbuf[(size_t)b * GN + f + 2 * H];
    float go = gbuf[(size_t)b * GN + f + 3 * H];
    float si = 1.f / (1.f + __expf(-gi));
    float sf = 1.f / (1.f + __expf(-gf));
    float so = 1.f / (1.f + __expf(-go));
    float cn = sf * c[(size_t)b * H + f] + si * tanhf(gg);
    float hn = so * tanhf(cn);
    c[(size_t)b * H + f] = cn;
    hbf[(size_t)b * H + f] = (__bf16)hn;   // recurrent input for next timestep
    hv[cnt++] = hn;
    s += hn; sq += hn * hn;
  }
  ssum[tid] = s; ssq[tid] = sq;
  __syncthreads();
  for (int off = 32; off > 0; off >>= 1) {
    if (tid < off) { ssum[tid] += ssum[tid + off]; ssq[tid] += ssq[tid + off]; }
    __syncthreads();
  }
  float mu  = ssum[0] * (1.f / BB);
  float var = ssq[0] * (1.f / BB) - mu * mu;   // biased variance
  float scale = rsqrtf(var + EPS) * gamma[f];
  float bet = beta[f];
  cnt = 0;
  for (int b = tid; b < BB; b += 64) {
    float a  = (hv[cnt++] - mu) * scale + bet;
    float am = a * mask[(size_t)b * H + f];
    abf[(size_t)b * H + f] = (__bf16)am;       // next-layer input
    if (isL3) {
      float* pp = &pooled[(size_t)b * H + f];
      *pp = fmaxf(*pp, am);
    }
  }
}

__device__ __forceinline__ unsigned hashu(unsigned x) {
  x ^= x >> 16; x *= 0x7feb352dU; x ^= x >> 15; x *= 0x846ca68bU; x ^= x >> 16; return x;
}

__global__ void init_state(float* __restrict__ c, __bf16* __restrict__ hbf,
                           __bf16* __restrict__ abf, float* __restrict__ masks,
                           float* __restrict__ pooled)
{
  int idx = blockIdx.x * blockDim.x + threadIdx.x;
  if (idx >= BP * H) return;
  for (int layer = 0; layer < 3; ++layer) {
    c  [(size_t)layer * BP * H + idx] = 0.f;
    hbf[(size_t)layer * BP * H + idx] = (__bf16)0.f;   // padded rows stay zero forever
    abf[(size_t)layer * BP * H + idx] = (__bf16)0.f;
  }
  if (idx < BB * H) {
    pooled[idx] = -3.0e38f;
    const float keep0 = 0.5f, keep12 = 0.7f;
    for (int layer = 0; layer < 3; ++layer) {
      float keep = (layer == 0) ? keep0 : keep12;
      unsigned u = hashu(idx * 2654435761u + layer * 0x9e3779b9u + 42u);
      float r = (u >> 8) * (1.0f / 16777216.0f);
      masks[(size_t)layer * BB * H + idx] = (r < keep) ? (1.f / keep) : 0.f;
    }
  }
}

// Gather embedding rows, cast to bf16, pad batch rows [BB,BP) with zeros.
__global__ __launch_bounds__(128) void embed_cast(const int* __restrict__ x,
                                                  const float* __restrict__ emb,
                                                  __bf16* __restrict__ embx)
{
  int t = blockIdx.x, b = blockIdx.y;
  size_t dst = ((size_t)t * BP + b) * H;
  if (b < BB) {
    int id = x[(size_t)b * TT + t];
    const float* src = emb + (size_t)id * H;
    for (int h = threadIdx.x; h < H; h += 128) embx[dst + h] = (__bf16)src[h];
  } else {
    for (int h = threadIdx.x; h < H; h += 128) embx[dst + h] = (__bf16)0.f;
  }
}

__global__ void cast_bf16(const float* __restrict__ src, __bf16* __restrict__ dst, int n)
{
  int i = blockIdx.x * blockDim.x + threadIdx.x;
  if (i < n) dst[i] = (__bf16)src[i];
}

__global__ __launch_bounds__(64) void fc_logits(const float* __restrict__ pooled,
                                                const float* __restrict__ fcW,
                                                const float* __restrict__ fcb,
                                                float* __restrict__ out)
{
  int b = blockIdx.x, tid = threadIdx.x;
  float s = 0.f;
  for (int h = tid; h < H; h += 64) s += pooled[(size_t)b * H + h] * fcW[h];
  __shared__ float red[64];
  red[tid] = s; __syncthreads();
  for (int off = 32; off > 0; off >>= 1) { if (tid < off) red[tid] += red[tid + off]; __syncthreads(); }
  if (tid == 0) out[1 + b] = red[0] + fcb[0];
}

__global__ __launch_bounds__(256) void bce_loss(const float* __restrict__ logits,
                                                const float* __restrict__ tgt,
                                                float* __restrict__ out0)
{
  int tid = threadIdx.x;
  float s = 0.f;
  for (int b = tid; b < BB; b += 256) {
    float z = logits[b], t = tgt[b];
    s += fmaxf(z, 0.f) - z * t + log1pf(__expf(-fabsf(z)));
  }
  __shared__ float red[256];
  red[tid] = s; __syncthreads();
  for (int off = 128; off > 0; off >>= 1) { if (tid < off) red[tid] += red[tid + off]; __syncthreads(); }
  if (tid == 0) out0[0] = red[0] * (1.f / BB);
}

extern "C" void kernel_launch(void* const* d_in, const int* in_sizes, int n_in,
                              void* d_out, int out_size, void* d_ws, size_t ws_size,
                              hipStream_t stream)
{
  (void)in_sizes; (void)n_in; (void)out_size; (void)ws_size;
  // Depth-first insertion-order flatten of setup_inputs():
  const int*   x   = (const int*)  d_in[0];
  const float* tgt = (const float*)d_in[1];
  const float* emb = (const float*)d_in[2];
  const float* Wih_f[3] = {(const float*)d_in[3],  (const float*)d_in[7],  (const float*)d_in[11]};
  const float* Whh_f[3] = {(const float*)d_in[4],  (const float*)d_in[8],  (const float*)d_in[12]};
  const float* bih[3]   = {(const float*)d_in[5],  (const float*)d_in[9],  (const float*)d_in[13]};
  const float* bhh[3]   = {(const float*)d_in[6],  (const float*)d_in[10], (const float*)d_in[14]};
  const float* gamma[3] = {(const float*)d_in[15], (const float*)d_in[17], (const float*)d_in[19]};
  const float* beta[3]  = {(const float*)d_in[16], (const float*)d_in[18], (const float*)d_in[20]};
  const float* fcW = (const float*)d_in[21];
  const float* fcb = (const float*)d_in[22];
  float* out = (float*)d_out;

  // Workspace carve (~45 MB total)
  char* ws = (char*)d_ws;
  size_t off = 0;
  auto alloc = [&](size_t bytes) -> char* {
    char* p = ws + off; off = (off + bytes + 255) & ~(size_t)255; return p;
  };
  __bf16* embx = (__bf16*)alloc((size_t)TT * BP * H * 2);
  __bf16* Wih_b[3]; __bf16* Whh_b[3];
  for (int l = 0; l < 3; ++l) {
    Wih_b[l] = (__bf16*)alloc((size_t)GN * H * 2);
    Whh_b[l] = (__bf16*)alloc((size_t)GN * H * 2);
  }
  __bf16* hbf  = (__bf16*)alloc((size_t)3 * BP * H * 2);
  __bf16* abf  = (__bf16*)alloc((size_t)3 * BP * H * 2);
  float*  c    = (float*) alloc((size_t)3 * BP * H * 4);
  float*  masks= (float*) alloc((size_t)3 * BB * H * 4);
  float*  gbuf = (float*) alloc((size_t)BP * GN * 4);
  float*  pooled=(float*) alloc((size_t)BB * H * 4);

  // Prologue: weight bf16 casts (also makes weights L2-resident), state init,
  // embedding gather+cast
  {
    int n = GN * H, blocks = (n + 255) / 256;
    for (int l = 0; l < 3; ++l) {
      cast_bf16<<<blocks, 256, 0, stream>>>(Wih_f[l], Wih_b[l], n);
      cast_bf16<<<blocks, 256, 0, stream>>>(Whh_f[l], Whh_b[l], n);
    }
  }
  init_state<<<(BP * H + 255) / 256, 256, 0, stream>>>(c, hbf, abf, masks, pooled);
  embed_cast<<<dim3(TT, BP), 128, 0, stream>>>(x, emb, embx);

  // Sequential recurrence: per timestep, per layer: WMMA gates GEMM -> fused cell+BN
  dim3 ggrid(2, GN / 16);
  for (int t = 0; t < TT; ++t) {
    const __bf16* ain = embx + (size_t)t * BP * H;
    for (int l = 0; l < 3; ++l) {
      __bf16* hb = hbf + (size_t)l * BP * H;
      __bf16* ab = abf + (size_t)l * BP * H;
      float*  cl = c   + (size_t)l * BP * H;
      const float* ml = masks + (size_t)l * BB * H;
      lstm_gates_wmma<<<ggrid, 256, 0, stream>>>(ain, Wih_b[l], hb, Whh_b[l],
                                                 bih[l], bhh[l], gbuf);
      lstm_cell_bn<<<H, 64, 0, stream>>>(gbuf, cl, hb, ab, gamma[l], beta[l],
                                         ml, pooled, (l == 2) ? 1 : 0);
      ain = ab;
    }
  }

  // Epilogue: FC head + BCE loss. d_out = [loss, logits(200)]
  fc_logits<<<BB, 64, 0, stream>>>(pooled, fcW, fcb, out);
  bce_loss<<<1, 256, 0, stream>>>(out + 1, tgt, out);
}
